// GCNLayer_63874753626440
// MI455X (gfx1250) — matrix operand order repaired and checked by reading, not compiled, not run on hardware
//
#include <hip/hip_runtime.h>
#include <hip/hip_bf16.h>

typedef __attribute__((ext_vector_type(16))) _Float16 v16h;
typedef __attribute__((ext_vector_type(8)))  float    v8f;

constexpr int N_NODES = 100000;
constexpr int N_EDGES = 1600000;
constexpr int D       = 128;
constexpr int N_PAD   = 100352;          // N_NODES rounded up to multiple of 1024

// workspace byte offsets (all 256B aligned)
constexpr size_t O_NSRC  = 0;                     // N_PAD f32 : deg_out -> norm_src
constexpr size_t O_NDST  = (size_t)N_PAD * 4;     // N_PAD f32 : deg_in  -> norm_dst
constexpr size_t O_SUM   = O_NDST + (size_t)N_PAD * 4;   // 128 f32
constexpr size_t O_SUMSQ = O_SUM + 512;                  // 128 f32
constexpr size_t O_WT    = O_SUMSQ + 512;                // 128*128 f16 (W transposed)
constexpr size_t O_AGG   = O_WT + (size_t)D * D * 2;     // N*128 f32
constexpr int    ZERO_WORDS = (int)(O_AGG / 4) + N_NODES * D;

__global__ void zero_kernel(float* __restrict__ p, int n) {
  for (int i = blockIdx.x * blockDim.x + threadIdx.x; i < n; i += gridDim.x * blockDim.x)
    p[i] = 0.0f;
}

__global__ __launch_bounds__(256) void degree_kernel(const int* __restrict__ src,
                                                     const int* __restrict__ dst,
                                                     float* __restrict__ deg_out,
                                                     float* __restrict__ deg_in) {
  int e = blockIdx.x * blockDim.x + threadIdx.x;
  if (e >= N_EDGES) return;
  atomicAdd(&deg_out[src[e]], 1.0f);
  atomicAdd(&deg_in[dst[e]], 1.0f);
}

__global__ __launch_bounds__(256) void norm_kernel(float* __restrict__ ns, float* __restrict__ nd) {
  int i = blockIdx.x * blockDim.x + threadIdx.x;
  if (i >= N_NODES) return;
  float a = ns[i];
  float b = nd[i];
  ns[i] = (a > 0.0f) ? rsqrtf(a) : 1.0f;
  nd[i] = (b > 0.0f) ? rsqrtf(b) : 1.0f;
}

// Wt[n][k] = (f16) W[k][n]  -> B fragments become K-contiguous
__global__ __launch_bounds__(256) void wt_kernel(const float* __restrict__ W, _Float16* __restrict__ Wt) {
  int i = blockIdx.x * blockDim.x + threadIdx.x;   // i = n*128 + k
  if (i >= D * D) return;
  int n = i >> 7, k = i & 127;
  Wt[i] = (_Float16)W[k * D + n];
}

// one wave per edge: agg[dst] += feat[src] * norm_src[src]
__global__ __launch_bounds__(256) void scatter_kernel(const float* __restrict__ feat,
                                                      const int* __restrict__ src,
                                                      const int* __restrict__ dst,
                                                      const float* __restrict__ nsrc,
                                                      float* __restrict__ agg) {
  int e    = blockIdx.x * 8 + (threadIdx.x >> 5);
  int lane = threadIdx.x & 31;
  if (e >= N_EDGES) return;
  int s = src[e], d = dst[e];
  float ns = nsrc[s];
  const float4 f = *(const float4*)(feat + (size_t)s * D + lane * 4);
  float* arow = agg + (size_t)d * D + lane * 4;
  atomicAdd(arow + 0, f.x * ns);
  atomicAdd(arow + 1, f.y * ns);
  atomicAdd(arow + 2, f.z * ns);
  atomicAdd(arow + 3, f.w * ns);
}

// 8 waves/block; block covers 16 node rows x 128 cols; wave w -> cols [16w,16w+16)
// K = 128 consumed as 4x v_wmma_f32_16x16x32_f16.
// norm_dst commutes through the row-linear GEMM, so it is applied in the
// epilogue (8 FMAs/wave via ds_bpermute) instead of 64 pk_muls in the A-load.
__global__ __launch_bounds__(256) void gemm_kernel(const float* __restrict__ agg,
                                                   const float* __restrict__ ndst,
                                                   const _Float16* __restrict__ Wt,
                                                   const float* __restrict__ bias,
                                                   float* __restrict__ out,
                                                   float* __restrict__ sum,
                                                   float* __restrict__ sumsq) {
  const int lane = threadIdx.x & 31;
  const int wave = threadIdx.x >> 5;
  const int half = lane >> 4;
  const int mr   = lane & 15;
  const int m0   = blockIdx.x * 16;
  const int row  = m0 + mr;            // A-matrix row this lane owns
  const int col  = wave * 16 + mr;     // B/C column this lane owns

  const float nd = ndst[row];          // lane mr holds norm_dst for row m0+mr
  const float* arow = agg + (size_t)row * D;
  const _Float16* wcol = Wt + (size_t)col * D;

  v8f c = {};
#pragma unroll
  for (int kb = 0; kb < 4; ++kb) {
    // A fragment (16-bit 16x32 layout): lane holds K = {8h..8h+7} U {16+8h..16+8h+7}
    const float4 f0 = *(const float4*)(arow + kb * 32 + half * 8);
    const float4 f1 = *(const float4*)(arow + kb * 32 + half * 8 + 4);
    const float4 f2 = *(const float4*)(arow + kb * 32 + 16 + half * 8);
    const float4 f3 = *(const float4*)(arow + kb * 32 + 16 + half * 8 + 4);
    v16h a;
    a[0]  = (_Float16)f0.x; a[1]  = (_Float16)f0.y;
    a[2]  = (_Float16)f0.z; a[3]  = (_Float16)f0.w;
    a[4]  = (_Float16)f1.x; a[5]  = (_Float16)f1.y;
    a[6]  = (_Float16)f1.z; a[7]  = (_Float16)f1.w;
    a[8]  = (_Float16)f2.x; a[9]  = (_Float16)f2.y;
    a[10] = (_Float16)f2.z; a[11] = (_Float16)f2.w;
    a[12] = (_Float16)f3.x; a[13] = (_Float16)f3.y;
    a[14] = (_Float16)f3.z; a[15] = (_Float16)f3.w;
    // B fragment: lane holds K = {8h..8h+7} U {16+8h..16+8h+7} for column `col`
    typedef __attribute__((ext_vector_type(8))) _Float16 v8h;
    const v8h b0 = *(const v8h*)(wcol + kb * 32 + half * 8);
    const v8h b1 = *(const v8h*)(wcol + kb * 32 + 16 + half * 8);
    v16h b = __builtin_shufflevector(b0, b1, 0, 1, 2, 3, 4, 5, 6, 7, 8, 9, 10, 11, 12, 13, 14, 15);
    c = __builtin_amdgcn_wmma_f32_16x16x32_f16(false, a, false, b, (short)0, c, false, false);
  }

  const float bv = bias[col];
  float s = 0.0f, s2 = 0.0f;
#pragma unroll
  for (int r = 0; r < 8; ++r) {
    int mloc = r + 8 * half;            // C/D layout: VGPR r -> M = r + 8*half
    // fetch norm_dst[m0+mloc] from the lane that owns it (lane mloc) — ds_bpermute
    float ndr = __int_as_float(__builtin_amdgcn_ds_bpermute(mloc * 4, __float_as_int(nd)));
    float v = fmaf(ndr, c[r], bv);
    v = v > 0.0f ? v : 0.0f;            // ReLU
    out[(size_t)(m0 + mloc) * D + col] = v;
    s += v; s2 += v * v;
  }
  atomicAdd(&sum[col], s);
  atomicAdd(&sumsq[col], s2);
}

// ---- JAX threefry2x32 (key = (0,42)), exact dropout mask ----
__device__ __forceinline__ unsigned rotl32(unsigned x, int r) { return (x << r) | (x >> (32 - r)); }

__device__ void threefry2x32(unsigned k0, unsigned k1, unsigned x0, unsigned x1,
                             unsigned* o0, unsigned* o1) {
  unsigned ks[3] = {k0, k1, k0 ^ k1 ^ 0x1BD11BDAu};
  x0 += ks[0]; x1 += ks[1];
  const int rotA[4] = {13, 15, 26, 6};
  const int rotB[4] = {17, 29, 16, 24};
#pragma unroll
  for (int i = 0; i < 5; ++i) {
    const int* rot = (i & 1) ? rotB : rotA;
#pragma unroll
    for (int j = 0; j < 4; ++j) { x0 += x1; x1 = rotl32(x1, rot[j]); x1 ^= x0; }
    x0 += ks[(i + 1) % 3];
    x1 += ks[(i + 2) % 3] + (unsigned)(i + 1);
  }
  *o0 = x0; *o1 = x1;
}

__global__ __launch_bounds__(256) void bn_drop_kernel(float* __restrict__ out,
                                                      const float* __restrict__ sum,
                                                      const float* __restrict__ sumsq,
                                                      const float* __restrict__ gamma,
                                                      const float* __restrict__ beta) {
  const unsigned TOT = (unsigned)N_NODES * D;
  const unsigned HALF = TOT / 2;
  unsigned idx = blockIdx.x * blockDim.x + threadIdx.x;
  if (idx >= TOT) return;
  int col = idx & 127;
  const float invN = 1.0f / (float)N_NODES;
  float mean = sum[col] * invN;
  float var  = sumsq[col] * invN - mean * mean;
  float v = (out[idx] - mean) * rsqrtf(var + 1e-5f) * gamma[col] + beta[col];

  unsigned bits, o0, o1;
  if (idx < HALF) { threefry2x32(0u, 42u, idx, idx + HALF, &o0, &o1); bits = o0; }
  else            { threefry2x32(0u, 42u, idx - HALF, idx, &o0, &o1); bits = o1; }
  float u = __uint_as_float((bits >> 9) | 0x3F800000u) - 1.0f;   // [0,1)
  out[idx] = (u < 0.5f) ? v * 2.0f : 0.0f;                        // p=0.5 keep, scale 1/(1-p)
}

extern "C" void kernel_launch(void* const* d_in, const int* in_sizes, int n_in,
                              void* d_out, int out_size, void* d_ws, size_t ws_size,
                              hipStream_t stream) {
  (void)in_sizes; (void)n_in; (void)out_size; (void)ws_size;
  const float* feat  = (const float*)d_in[0];
  const int*   src   = (const int*)d_in[1];
  const int*   dst   = (const int*)d_in[2];
  const float* W     = (const float*)d_in[3];
  const float* bias  = (const float*)d_in[4];
  const float* gamma = (const float*)d_in[5];
  const float* beta  = (const float*)d_in[6];
  float* out = (float*)d_out;

  char* ws = (char*)d_ws;
  float*     nsrc  = (float*)(ws + O_NSRC);
  float*     ndst  = (float*)(ws + O_NDST);
  float*     sum   = (float*)(ws + O_SUM);
  float*     sumsq = (float*)(ws + O_SUMSQ);
  _Float16*  Wt    = (_Float16*)(ws + O_WT);
  float*     agg   = (float*)(ws + O_AGG);

  zero_kernel<<<2048, 256, 0, stream>>>((float*)d_ws, ZERO_WORDS);
  degree_kernel<<<(N_EDGES + 255) / 256, 256, 0, stream>>>(src, dst, nsrc, ndst);
  norm_kernel<<<(N_NODES + 255) / 256, 256, 0, stream>>>(nsrc, ndst);
  wt_kernel<<<(D * D + 255) / 256, 256, 0, stream>>>(W, Wt);
  scatter_kernel<<<N_EDGES / 8, 256, 0, stream>>>(feat, src, dst, nsrc, agg);
  gemm_kernel<<<N_NODES / 16, 256, 0, stream>>>(agg, ndst, Wt, bias, out, sum, sumsq);
  bn_drop_kernel<<<(N_NODES * D + 255) / 256, 256, 0, stream>>>(out, sum, sumsq, gamma, beta);
}
